// EdgeDetectionModule_50603304681496
// MI455X (gfx1250) — matrix operand order repaired and checked by reading, not compiled, not run on hardware
//
#include <hip/hip_runtime.h>
#include <hip/hip_bf16.h>
#include <cmath>

// ---------------------------------------------------------------------------
// EdgeDetection pipeline for MI455X (gfx1250, wave32, WMMA + TDM).
// Convs -> implicit-GEMM with v_wmma_f32_16x16x32_f16; input tile staged into
// LDS by the Tensor Data Mover (tensor_load_to_lds + s_wait_tensorcnt).
// Canny -> streaming kernels; hysteresis fixpoint approximated by 32 dilation
// passes (deterministic, graph-capture safe, L2-resident at 16.8MB/plane).
// ---------------------------------------------------------------------------

typedef __attribute__((ext_vector_type(16))) _Float16 v16h;
typedef __attribute__((ext_vector_type(8)))  float    v8f;
typedef __attribute__((ext_vector_type(4)))  unsigned int u32x4;
typedef __attribute__((ext_vector_type(4)))  int      i32x4;
typedef __attribute__((ext_vector_type(8)))  int      i32x8;

#define HH 512
#define WW 512

__device__ __forceinline__ float ldz(const float* __restrict__ p, int y, int x,
                                     int H, int W) {
  return (x >= 0 && x < W && y >= 0 && y < H) ? p[(size_t)y * W + x] : 0.0f;
}

// ---------------------------------------------------------------------------
// WMMA implicit-GEMM 3x3 SAME conv, NCHW fp32 in/out, f16 MACs, f32 accum.
// Block = 256 threads = 8 waves; each wave computes a 16-pixel x COUT tile of
// one output row via K = CIN*9 (zero-padded to 32*KCH) chunks of WMMA.
// Input tile (CIN x 3 x 130 f32) is DMA'd to LDS by the TDM from wave 0;
// image-border halo cells are zero-fixed after s_wait_tensorcnt.
// ACT: 0 none, 1 leaky-relu(0.2), 2 sigmoid.
// ---------------------------------------------------------------------------
template <int CIN, int COUT, int ACT>
__global__ __launch_bounds__(256)
void conv3x3_wmma(const float* __restrict__ in, const float* __restrict__ wgt,
                  const float* __restrict__ bias, float* __restrict__ out,
                  int B, int H, int W) {
  constexpr int K    = CIN * 9;
  constexpr int KCH  = (K + 31) / 32;   // 32-wide K chunks
  constexpr int TILE = 128;             // pixels per block
  constexpr int LW   = TILE + 2;        // 130: haloed row width (TDM-contiguous)

  __shared__ float    s_in[(CIN * 3 + 1) * LW];  // tile f32 + one zero row
  __shared__ _Float16 s_w[KCH * 32 * 16];        // B-matrix-ready weights
  __shared__ short    s_aoff[KCH * 32];          // k -> LDS row/col offset

  const int tid    = threadIdx.x;
  const int tilesX = W / TILE;
  int bid = blockIdx.x;
  const int tx = bid % tilesX; bid /= tilesX;
  const int y  = bid % H;      bid /= H;
  const int b  = bid;
  const int x0 = tx * TILE;

  // ---- TDM: async DMA of haloed input tile into LDS (issued by wave 0) ----
  // 3D tile: X = LW cols (contiguous), Y = 3 rows (stride W), Z = CIN (stride
  // H*W). Border cells map to out-of-tensor addresses; they are overwritten
  // with zeros below before use, implementing SAME zero padding.
  if (tid < 32) {
    const long long startEl =
        ((long long)(b * CIN) * H + (long long)(y - 1)) * W + (x0 - 1);
    const unsigned long long gaddr =
        (unsigned long long)(uintptr_t)in + (unsigned long long)(4ll * startEl);
    const unsigned lds_base = (unsigned)(uintptr_t)(void*)&s_in[0];

    u32x4 g0 = {0u, 0u, 0u, 0u};
    g0[0] = 1u;                                   // count=1, user descriptor
    g0[1] = lds_base;                             // lds_addr (bytes)
    g0[2] = (unsigned)gaddr;                      // global_addr[31:0]
    g0[3] = ((unsigned)(gaddr >> 32) & 0x01FFFFFFu) | (2u << 30);  // [56:32]|type=2

    const unsigned t1s = (unsigned)(H * W);       // tensor_dim1_stride (elems)
    i32x8 g1 = {0, 0, 0, 0, 0, 0, 0, 0};
    g1[0] = (2 << 16);                            // data_size = 4 bytes
    g1[1] = (int)(((unsigned)W & 0xFFFFu) << 16); // tensor_dim0[15:0]
    g1[2] = (int)((((unsigned)W >> 16) & 0xFFFFu) |
                  (((unsigned)H & 0xFFFFu) << 16));          // td0 hi | td1 lo
    g1[3] = (int)((((unsigned)H >> 16) & 0xFFFFu) |
                  ((unsigned)LW << 16));                     // td1 hi | tile_dim0
    g1[4] = (int)(3u | ((unsigned)CIN << 16));    // tile_dim1=3 | tile_dim2=CIN
    g1[5] = W;                                    // tensor_dim0_stride[31:0]
    g1[6] = (int)((t1s & 0xFFFFu) << 16);         // t0s[47:32]=0 | t1s[15:0]
    g1[7] = (int)(t1s >> 16);                     // t1s[47:16]
    i32x4 g2 = {CIN, 0, 0, 0};                    // tensor_dim2
    i32x4 g3 = {0, 0, 0, 0};
#if defined(__clang_major__) && (__clang_major__ >= 23)
    i32x8 g4 = {0, 0, 0, 0, 0, 0, 0, 0};
    __builtin_amdgcn_tensor_load_to_lds(g0, g1, g2, g3, g4, 0);
#else
    __builtin_amdgcn_tensor_load_to_lds(g0, g1, g2, g3, 0);
#endif
    __builtin_amdgcn_s_wait_tensorcnt((short)0);
  }

  // Stage weights: s_w[k][n] = wgt[n*K + k], zero-padded in both k and n.
  for (int i = tid; i < KCH * 32 * 16; i += 256) {
    const int k = i >> 4, n = i & 15;
    float v = 0.0f;
    if (k < K && n < COUT) v = wgt[n * K + k];
    s_w[i] = (_Float16)v;
  }
  // k -> A-gather offset table (out-of-range K points at the zero row).
  for (int i = tid; i < KCH * 32; i += 256) {
    short off;
    if (i < K) {
      const int ci = i / 9, rr = (i % 9) / 3, cc = i % 3;
      off = (short)((ci * 3 + rr) * LW + cc);
    } else {
      off = (short)(CIN * 3 * LW);
    }
    s_aoff[i] = off;
  }
  __syncthreads();   // TDM data + tables visible

  // Zero row (for padded K) and SAME-padding border fix-up.
  for (int i = tid; i < LW; i += 256) s_in[CIN * 3 * LW + i] = 0.0f;
  if (y == 0)
    for (int i = tid; i < CIN * LW; i += 256)
      s_in[(i / LW) * 3 * LW + (i % LW)] = 0.0f;
  if (y == H - 1)
    for (int i = tid; i < CIN * LW; i += 256)
      s_in[(i / LW) * 3 * LW + 2 * LW + (i % LW)] = 0.0f;
  if (x0 == 0)
    for (int i = tid; i < CIN * 3; i += 256) s_in[i * LW] = 0.0f;
  if (x0 + TILE == W)
    for (int i = tid; i < CIN * 3; i += 256) s_in[i * LW + (LW - 1)] = 0.0f;
  __syncthreads();

  const int wid  = tid >> 5;        // wave 0..7 -> 16-pixel sub-tile
  const int lane = tid & 31;
  const int m    = lane & 15;       // A row (pixel) / B col (cout) / D col
  const int hi   = lane >> 4;
  const int abase = wid * 16 + m;   // column base for this lane's pixel

  v8f acc = {};
#pragma unroll
  for (int kc = 0; kc < KCH; ++kc) {
    v16h a, bm;
#pragma unroll
    for (int h = 0; h < 16; ++h) {
      // A 16x32 layout: lane holds row m; half h -> K = (h<8?h:h+8) + 8*hi.
      const int ka = kc * 32 + (h < 8 ? h : h + 8) + hi * 8;
      a[h] = (_Float16)s_in[(int)s_aoff[ka] + abase];
      // B 32x16 layout: lane holds col n=m; half h -> K = (h<8?h:h+8) + 8*hi.
      const int kb = kc * 32 + (h < 8 ? h : h + 8) + hi * 8;
      bm[h] = s_w[kb * 16 + m];
    }
    acc = __builtin_amdgcn_wmma_f32_16x16x32_f16(
        /*neg_a=*/false, a, /*neg_b=*/false, bm,
        /*c_mod=*/(short)0, acc, /*reuse_a=*/false, /*reuse_b=*/false);
  }

  // D layout: lane -> N=co, VGPR v -> M = v + 8*hi (pixel within wave tile).
  const int co = m;
  if (co < COUT) {
    const float bv = bias[co];
#pragma unroll
    for (int v = 0; v < 8; ++v) {
      const int px = x0 + wid * 16 + v + hi * 8;
      float val = acc[v] + bv;
      if (ACT == 1) val = (val >= 0.0f) ? val : 0.2f * val;
      else if (ACT == 2) val = 1.0f / (1.0f + __expf(-val));
      out[((size_t)(b * COUT + co) * H + y) * W + px] = val;
    }
  }
}

// ----------------------------- streaming kernels ---------------------------

__global__ void k_gray(const float* __restrict__ x, float* __restrict__ g,
                       int N, int HW) {
  int i = blockIdx.x * blockDim.x + threadIdx.x;
  if (i >= N) return;
  const int b = i / HW, p = i % HW;
  const float* xb = x + (size_t)b * 3 * HW;
  g[i] = 0.299f * xb[p] + 0.587f * xb[HW + p] + 0.114f * xb[2 * HW + p];
}

__global__ void k_zero(float* __restrict__ d, int N) {
  int i = blockIdx.x * blockDim.x + threadIdx.x;
  if (i < N) d[i] = 0.0f;
}

__global__ void k_blur_h(const float* __restrict__ in, float* __restrict__ out,
                         int N, int W, float k0, float k1, float k2) {
  int i = blockIdx.x * blockDim.x + threadIdx.x;
  if (i >= N) return;
  const int x = i % W;
  const float* row = in + (size_t)(i - x);
  float v = k2 * row[x];
  v += k1 * ((x >= 1 ? row[x - 1] : 0.0f) + (x + 1 < W ? row[x + 1] : 0.0f));
  v += k0 * ((x >= 2 ? row[x - 2] : 0.0f) + (x + 2 < W ? row[x + 2] : 0.0f));
  out[i] = v;
}

__global__ void k_blur_v(const float* __restrict__ in, float* __restrict__ out,
                         int N, int H, int W, float k0, float k1, float k2) {
  int i = blockIdx.x * blockDim.x + threadIdx.x;
  if (i >= N) return;
  const int x = i % W;
  const int t = i / W;
  const int y = t % H;
  const int b = t / H;
  const float* p = in + (size_t)b * H * W;
  float v = k2 * p[(size_t)y * W + x];
  v += k1 * (ldz(p, y - 1, x, H, W) + ldz(p, y + 1, x, H, W));
  v += k0 * (ldz(p, y - 2, x, H, W) + ldz(p, y + 2, x, H, W));
  out[i] = v;
}

__global__ void k_sobel(const float* __restrict__ in, float* __restrict__ mag,
                        float* __restrict__ qd, int N, int H, int W) {
  int i = blockIdx.x * blockDim.x + threadIdx.x;
  if (i >= N) return;
  const int x = i % W;
  const int t = i / W;
  const int y = t % H;
  const int b = t / H;
  const float* p = in + (size_t)b * H * W;
  const float p00 = ldz(p, y - 1, x - 1, H, W), p01 = ldz(p, y - 1, x, H, W),
              p02 = ldz(p, y - 1, x + 1, H, W);
  const float p10 = ldz(p, y, x - 1, H, W), p12 = ldz(p, y, x + 1, H, W);
  const float p20 = ldz(p, y + 1, x - 1, H, W), p21 = ldz(p, y + 1, x, H, W),
              p22 = ldz(p, y + 1, x + 1, H, W);
  const float gx = (p02 - p00) + 2.0f * (p12 - p10) + (p22 - p20);
  const float gy = (p20 - p00) + 2.0f * (p21 - p01) + (p22 - p02);
  const float m = sqrtf(gx * gx + gy * gy + 1e-12f);
  const float ang = atan2f(gy, gx);
  int q = (int)rintf(ang * (4.0f / 3.14159265358979323846f));
  q = ((q % 4) + 4) % 4;
  mag[i] = m;
  qd[i] = (float)q;
}

__global__ void k_nms(const float* __restrict__ mag, const float* __restrict__ qd,
                      float* __restrict__ strong, float* __restrict__ weak,
                      int N, int H, int W, float low, float high) {
  int i = blockIdx.x * blockDim.x + threadIdx.x;
  if (i >= N) return;
  const int x = i % W;
  const int t = i / W;
  const int y = t % H;
  const int b = t / H;
  const float* p = mag + (size_t)b * H * W;
  const float m = p[(size_t)y * W + x];
  const int q = (int)qd[i];
  float na, nb;
  if (q == 0)      { na = ldz(p, y,     x + 1, H, W); nb = ldz(p, y,     x - 1, H, W); }
  else if (q == 1) { na = ldz(p, y - 1, x + 1, H, W); nb = ldz(p, y + 1, x - 1, H, W); }
  else if (q == 2) { na = ldz(p, y - 1, x,     H, W); nb = ldz(p, y + 1, x,     H, W); }
  else             { na = ldz(p, y - 1, x - 1, H, W); nb = ldz(p, y + 1, x + 1, H, W); }
  const float nms = ((m >= na) && (m >= nb)) ? m : 0.0f;
  strong[i] = (nms > high) ? 1.0f : 0.0f;
  weak[i]   = ((nms > low) && (nms <= high)) ? 1.0f : 0.0f;
}

__global__ void k_hyst(const float* __restrict__ e, const float* __restrict__ weak,
                       float* __restrict__ eo, int N, int H, int W) {
  int i = blockIdx.x * blockDim.x + threadIdx.x;
  if (i >= N) return;
  const int x = i % W;
  const int t = i / W;
  const int y = t % H;
  const int b = t / H;
  const float* p = e + (size_t)b * H * W;
  if (y + 8 < H) __builtin_prefetch(&p[(size_t)(y + 8) * W + x], 0, 1);
  float d = 0.0f;
#pragma unroll
  for (int dy = -1; dy <= 1; ++dy)
#pragma unroll
    for (int dx = -1; dx <= 1; ++dx)
      d = fmaxf(d, ldz(p, y + dy, x + dx, H, W));
  eo[i] = fmaxf(e[i], weak[i] * (d > 0.0f ? 1.0f : 0.0f));
}

__global__ void k_maxinto(float* __restrict__ dst, const float* __restrict__ src,
                          int N) {
  int i = blockIdx.x * blockDim.x + threadIdx.x;
  if (i < N) dst[i] = fmaxf(dst[i], src[i]);
}

__global__ void k_pack2(const float* __restrict__ c0, const float* __restrict__ c1,
                        float* __restrict__ out, int N, int HW) {
  int i = blockIdx.x * blockDim.x + threadIdx.x;
  if (i >= N) return;
  const int b = i / HW, p = i % HW;
  out[(size_t)b * 2 * HW + p]      = c0[i];
  out[(size_t)b * 2 * HW + HW + p] = c1[i];
}

// d_out layout: [edge_map S][edge_features 2S][combined S][deep S], S=B*HW.
__global__ void k_final(const float* __restrict__ refined,
                        const float* __restrict__ deep,
                        const float* __restrict__ combined,
                        const float* __restrict__ mask,
                        float* __restrict__ out, int N, int HW) {
  int i = blockIdx.x * blockDim.x + threadIdx.x;
  if (i >= N) return;
  const int b = i / HW, p = i % HW;
  const size_t S = (size_t)N;
  const float mk = mask[i];
  const float dv = deep[i];
  const float em = refined[i] * (1.0f - mk) + dv * mk * 0.5f;
  out[i] = em;                                   // edge_map
  out[S + (size_t)b * 2 * HW + p]      = em;     // edge_features ch0
  out[S + (size_t)b * 2 * HW + HW + p] = dv;     // edge_features ch1
  out[3 * S + i] = combined[i];                  // combined_canny
  out[4 * S + i] = dv;                           // deep_edges
}

// ----------------------------------- host ----------------------------------

static inline void gauss3(float sigma, float& k0, float& k1, float& k2) {
  float w[5], s = 0.0f;
  for (int i = 0; i < 5; ++i) {
    const float r = (float)i - 2.0f;
    w[i] = std::exp(-(r * r) / (2.0f * sigma * sigma));
    s += w[i];
  }
  k0 = w[0] / s; k1 = w[1] / s; k2 = w[2] / s;
}

extern "C" void kernel_launch(void* const* d_in, const int* in_sizes, int n_in,
                              void* d_out, int out_size, void* d_ws, size_t ws_size,
                              hipStream_t stream) {
  (void)n_in; (void)out_size; (void)ws_size;
  const int H = HH, W = WW, HW = H * W;
  const int B = in_sizes[1] / HW;           // mask is (B,1,H,W)
  const int N = B * HW;

  const float* x    = (const float*)d_in[0];
  const float* mask = (const float*)d_in[1];
  const float* wd1 = (const float*)d_in[2];  const float* bd1 = (const float*)d_in[3];
  const float* wd2 = (const float*)d_in[4];  const float* bd2 = (const float*)d_in[5];
  const float* wd3 = (const float*)d_in[6];  const float* bd3 = (const float*)d_in[7];
  const float* wd4 = (const float*)d_in[8];  const float* bd4 = (const float*)d_in[9];
  const float* wr1 = (const float*)d_in[10]; const float* br1 = (const float*)d_in[11];
  const float* wr2 = (const float*)d_in[12]; const float* br2 = (const float*)d_in[13];
  const float* wr3 = (const float*)d_in[14]; const float* br3 = (const float*)d_in[15];

  float* ws = (float*)d_ws;
  const size_t S = (size_t)N;
  float* combined = ws + 0 * S;
  float* deep     = ws + 1 * S;
  float* gray     = ws + 2 * S;
  float* t1       = ws + 3 * S;
  float* t2       = ws + 4 * S;
  float* mag      = ws + 5 * S;
  float* qd       = ws + 6 * S;
  float* weak     = ws + 7 * S;
  float* e0       = ws + 8 * S;
  float* e1       = ws + 9 * S;
  float* h1       = ws + 10 * S;            // 16 planes
  float* h2       = ws + 26 * S;            // 16 planes
  float* h3       = ws + 10 * S;            // 8 planes, reuses h1
  float* ce       = ws + 26 * S;            // 2 planes, reuses h2
  float* r1       = ws + 28 * S;            // 8 planes
  float* r2       = ws + 36 * S;            // 4 planes
  float* refined  = ws + 2 * S;             // reuses gray

  const dim3 blk(256);
  const dim3 grd((N + 255) / 256);
  const dim3 grdConv((W / 128) * H * B);

  const float LOW = 0.1f, HIGH = 0.2f;
  const float sigmas[3] = {0.5f, 1.0f, 2.0f};
  float i0, i1, i2; gauss3(1.0f, i0, i1, i2);   // canny-internal blur

  k_gray<<<grd, blk, 0, stream>>>(x, gray, N, HW);
  k_zero<<<grd, blk, 0, stream>>>(combined, N);

  for (int si = 0; si < 3; ++si) {
    float s0, s1, s2; gauss3(sigmas[si], s0, s1, s2);
    // outer blur(gray, sigma)
    k_blur_h<<<grd, blk, 0, stream>>>(gray, t1, N, W, s0, s1, s2);
    k_blur_v<<<grd, blk, 0, stream>>>(t1, t2, N, H, W, s0, s1, s2);
    // canny-internal blur sigma=1
    k_blur_h<<<grd, blk, 0, stream>>>(t2, t1, N, W, i0, i1, i2);
    k_blur_v<<<grd, blk, 0, stream>>>(t1, t2, N, H, W, i0, i1, i2);
    k_sobel<<<grd, blk, 0, stream>>>(t2, mag, qd, N, H, W);
    k_nms<<<grd, blk, 0, stream>>>(mag, qd, e0, weak, N, H, W, LOW, HIGH);
    // fixed-count hysteresis propagation (even count -> result lands in e0)
    for (int it = 0; it < 16; ++it) {
      k_hyst<<<grd, blk, 0, stream>>>(e0, weak, e1, N, H, W);
      k_hyst<<<grd, blk, 0, stream>>>(e1, weak, e0, N, H, W);
    }
    k_maxinto<<<grd, blk, 0, stream>>>(combined, e0, N);
  }

  // deep stack (WMMA, TDM-staged)
  conv3x3_wmma<3, 16, 1><<<grdConv, blk, 0, stream>>>(x,  wd1, bd1, h1, B, H, W);
  conv3x3_wmma<16, 16, 1><<<grdConv, blk, 0, stream>>>(h1, wd2, bd2, h2, B, H, W);
  conv3x3_wmma<16, 8, 1><<<grdConv, blk, 0, stream>>>(h2, wd3, bd3, h3, B, H, W);
  conv3x3_wmma<8, 1, 2><<<grdConv, blk, 0, stream>>>(h3, wd4, bd4, deep, B, H, W);

  // refinement stack (WMMA, TDM-staged)
  k_pack2<<<grd, blk, 0, stream>>>(combined, deep, ce, N, HW);
  conv3x3_wmma<2, 8, 1><<<grdConv, blk, 0, stream>>>(ce, wr1, br1, r1, B, H, W);
  conv3x3_wmma<8, 4, 1><<<grdConv, blk, 0, stream>>>(r1, wr2, br2, r2, B, H, W);
  conv3x3_wmma<4, 1, 2><<<grdConv, blk, 0, stream>>>(r2, wr3, br3, refined, B, H, W);

  k_final<<<grd, blk, 0, stream>>>(refined, deep, combined, mask,
                                   (float*)d_out, N, HW);
}
